// GraphEncoder_33749853012495
// MI455X (gfx1250) — matrix operand order repaired and checked by reading, not compiled, not run on hardware
//
#include <hip/hip_runtime.h>
#include <hip/hip_bf16.h>
#include <cstdint>

typedef __attribute__((ext_vector_type(16))) __bf16 v16bf;
typedef __attribute__((ext_vector_type(8)))  float  v8f;

#define DFEAT 128
#define TSTRIDE 132   // padded LDS row stride (floats) to avoid bank conflicts

// ---------------------------------------------------------------------------
// Split fp32 weight matrix into bf16 hi/lo tables (W = hi + lo to ~2^-17 rel)
// ---------------------------------------------------------------------------
__global__ void k_wsplit(const float* __restrict__ W,
                         __bf16* __restrict__ hi, __bf16* __restrict__ lo, int n) {
    int i = blockIdx.x * blockDim.x + threadIdx.x;
    if (i >= n) return;
    float w = W[i];
    __bf16 h = (__bf16)w;
    hi[i] = h;
    lo[i] = (__bf16)(w - (float)h);
}

// ---------------------------------------------------------------------------
// Degree count (incoming edges at dst) and dinv = rsqrt(deg + 1)
// ---------------------------------------------------------------------------
__global__ void k_deg(const int* __restrict__ dst, float* __restrict__ deg, int E) {
    int e = blockIdx.x * blockDim.x + threadIdx.x;
    if (e < E) atomicAdd(&deg[dst[e]], 1.0f);
}

__global__ void k_dinv(float* __restrict__ deg, int N) {
    int i = blockIdx.x * blockDim.x + threadIdx.x;
    if (i < N) deg[i] = rsqrtf(deg[i] + 1.0f);
}

// ---------------------------------------------------------------------------
// H[N x 128] = X[N x 128] @ W[128 x 128] via v_wmma_f32_16x16x32_bf16.
// Split-precision: X=Xhi+Xlo, W=Whi+Wlo; acc += Xhi*Whi + Xhi*Wlo + Xlo*Whi.
// Block = 256 threads (8 waves); block handles 16 rows, wave w handles cols
// [16w, 16w+16). A panel (16x128 f32) staged to LDS with async-to-LDS DMA.
// ---------------------------------------------------------------------------
__global__ __launch_bounds__(256)
void k_gemm_bf16x3(const float* __restrict__ X,
                   const __bf16* __restrict__ Whi, const __bf16* __restrict__ Wlo,
                   float* __restrict__ Hout, int N) {
    __shared__ float tile[16 * TSTRIDE];

    const int t  = threadIdx.x;
    const int m0 = blockIdx.x * 16;

    // Stage 16 rows x 128 f32 = 512 x 16B chunks via async DMA to LDS.
    for (int c = t; c < 512; c += 256) {
        int row = c >> 5;              // 0..15
        int off = (c & 31) << 2;       // float offset within row, 0..124
        int gr  = m0 + row; if (gr >= N) gr = N - 1;
        const float* gp = X + (size_t)gr * DFEAT + off;
        unsigned lds = (unsigned)(uintptr_t)(&tile[row * TSTRIDE + off]);
        asm volatile("global_load_async_to_lds_b128 %0, %1, off"
                     :: "v"(lds), "v"(gp) : "memory");
    }
    asm volatile("s_wait_asynccnt 0x0" ::: "memory");
    __syncthreads();

    const int wave  = t >> 5;
    const int lane  = t & 31;
    const int n0    = wave * 16;
    const int Mloc  = lane & 15;
    const int halfw = lane >> 4;

    v8f acc = {0.f, 0.f, 0.f, 0.f, 0.f, 0.f, 0.f, 0.f};

    #pragma unroll
    for (int kt = 0; kt < 4; ++kt) {
        // A operand (16x32 bf16): lane holds row Mloc; elements 0..7 are
        // K = kt*32 + halfw*8 + e, elements 8..15 are K = kt*32+16+halfw*8+e.
        const float* p = &tile[Mloc * TSTRIDE + kt * 32 + halfw * 8];
        v16bf ahi, alo;
        #pragma unroll
        for (int e = 0; e < 8; ++e) {
            float f1 = p[e];
            float f2 = p[16 + e];
            __bf16 h1 = (__bf16)f1;
            __bf16 h2 = (__bf16)f2;
            ahi[e]     = h1;  alo[e]     = (__bf16)(f1 - (float)h1);
            ahi[8 + e] = h2;  alo[8 + e] = (__bf16)(f2 - (float)h2);
        }
        // B operand (32x16 bf16): lane holds column N = n0+Mloc; elements 0..7 are
        // K = kt*32 + halfw*8 + e, elements 8..15 are K = kt*32+16+halfw*8+e.
        v16bf bhi, blo;
        #pragma unroll
        for (int e = 0; e < 8; ++e) {
            const size_t i1 = (size_t)(kt * 32 + halfw * 8 + e) * DFEAT + n0 + Mloc;
            const size_t i2 = i1 + (size_t)16 * DFEAT;
            bhi[e] = Whi[i1];  blo[e] = Wlo[i1];
            bhi[8 + e] = Whi[i2];  blo[8 + e] = Wlo[i2];
        }

        acc = __builtin_amdgcn_wmma_f32_16x16x32_bf16(false, ahi, false, bhi,
                                                      (short)0, acc, false, false);
        acc = __builtin_amdgcn_wmma_f32_16x16x32_bf16(false, ahi, false, blo,
                                                      (short)0, acc, false, false);
        acc = __builtin_amdgcn_wmma_f32_16x16x32_bf16(false, alo, false, bhi,
                                                      (short)0, acc, false, false);
    }

    // C/D layout: VGPR r, lanes 0-15 -> M = r, lanes 16-31 -> M = r + 8.
    const int col = n0 + (lane & 15);
    #pragma unroll
    for (int r = 0; r < 8; ++r) {
        int row = m0 + r + 8 * halfw;
        if (row < N) Hout[(size_t)row * DFEAT + col] = acc[r];
    }
}

// ---------------------------------------------------------------------------
// Edge scatter: agg[dst] += h[src] * dinv[src]*dinv[dst].  One wave per edge,
// each lane moves a float4 (4 x global_atomic_add_f32, L2-resident).
// ---------------------------------------------------------------------------
__global__ __launch_bounds__(256)
void k_scatter(const float* __restrict__ h, const int* __restrict__ src,
               const int* __restrict__ dst, const float* __restrict__ dinv,
               float* __restrict__ agg, int E) {
    int e = blockIdx.x * 8 + (threadIdx.x >> 5);
    if (e >= E) return;
    int lane = threadIdx.x & 31;
    int s = src[e], d = dst[e];
    float nrm = dinv[s] * dinv[d];
    const float4 v = *(const float4*)(h + (size_t)s * DFEAT + lane * 4);
    float* o = agg + (size_t)d * DFEAT + lane * 4;
    atomicAdd(o + 0, v.x * nrm);
    atomicAdd(o + 1, v.y * nrm);
    atomicAdd(o + 2, v.z * nrm);
    atomicAdd(o + 3, v.w * nrm);
}

// ---------------------------------------------------------------------------
// agg = relu(agg + h * dinv^2 + b)   (self-loop + bias + activation)
// ---------------------------------------------------------------------------
__global__ void k_fuse(float* __restrict__ agg, const float* __restrict__ h,
                       const float* __restrict__ dinv, const float* __restrict__ b,
                       int N) {
    int idx = blockIdx.x * blockDim.x + threadIdx.x;   // N*32 float4 units
    if (idx >= N * 32) return;
    int node = idx >> 5, ch = idx & 31;
    float d2 = dinv[node]; d2 *= d2;
    float4 a  = *((const float4*)agg + idx);
    float4 hh = *((const float4*)h + idx);
    float4 bb = *((const float4*)b + ch);
    float4 r;
    r.x = fmaxf(a.x + hh.x * d2 + bb.x, 0.f);
    r.y = fmaxf(a.y + hh.y * d2 + bb.y, 0.f);
    r.z = fmaxf(a.z + hh.z * d2 + bb.z, 0.f);
    r.w = fmaxf(a.w + hh.w * d2 + bb.w, 0.f);
    *((float4*)agg + idx) = r;
}

// ---------------------------------------------------------------------------
// Mean pool over batch segments
// ---------------------------------------------------------------------------
__global__ void k_pool(const float* __restrict__ h, const int* __restrict__ batch,
                       float* __restrict__ out, float* __restrict__ cnt, int N) {
    int idx = blockIdx.x * blockDim.x + threadIdx.x;   // N*32 float4 units
    if (idx >= N * 32) return;
    int node = idx >> 5, ch = idx & 31;
    int g = batch[node];
    float4 v = *((const float4*)h + idx);
    float* o = out + (size_t)g * DFEAT + ch * 4;
    atomicAdd(o + 0, v.x);
    atomicAdd(o + 1, v.y);
    atomicAdd(o + 2, v.z);
    atomicAdd(o + 3, v.w);
    if (ch == 0) atomicAdd(&cnt[g], 1.0f);
}

__global__ void k_final(float* __restrict__ out, const float* __restrict__ cnt, int n) {
    int i = blockIdx.x * blockDim.x + threadIdx.x;
    if (i < n) out[i] /= fmaxf(cnt[i >> 7], 1.0f);
}

// ---------------------------------------------------------------------------
extern "C" void kernel_launch(void* const* d_in, const int* in_sizes, int n_in,
                              void* d_out, int out_size, void* d_ws, size_t ws_size,
                              hipStream_t stream) {
    const float* x   = (const float*)d_in[0];
    const float* W1  = (const float*)d_in[1];
    const float* b1  = (const float*)d_in[2];
    const float* W2  = (const float*)d_in[3];
    const float* b2  = (const float*)d_in[4];
    const int* edge  = (const int*)d_in[5];
    const int* batch = (const int*)d_in[6];
    float* out = (float*)d_out;

    const int N = in_sizes[0] / DFEAT;
    const int E = in_sizes[5] / 2;
    const int G = out_size / DFEAT;
    const int* src = edge;
    const int* dst = edge + E;

    // workspace carve-up (256B aligned slabs)
    char* ws = (char*)d_ws;
    size_t o = 0;
    auto alloc = [&](size_t bytes) {
        char* p = ws + o;
        o = (o + bytes + 255) & ~(size_t)255;
        return p;
    };
    __bf16* Whi1 = (__bf16*)alloc((size_t)DFEAT * DFEAT * 2);
    __bf16* Wlo1 = (__bf16*)alloc((size_t)DFEAT * DFEAT * 2);
    __bf16* Whi2 = (__bf16*)alloc((size_t)DFEAT * DFEAT * 2);
    __bf16* Wlo2 = (__bf16*)alloc((size_t)DFEAT * DFEAT * 2);
    float*  dinv = (float*)alloc((size_t)N * 4);
    float*  bufH = (float*)alloc((size_t)N * DFEAT * 4);
    float*  bufA = (float*)alloc((size_t)N * DFEAT * 4);
    float*  cnt  = (float*)alloc((size_t)G * 4);
    (void)ws_size; (void)n_in;

    const size_t nd_bytes = (size_t)N * DFEAT * 4;

    // weight splits
    k_wsplit<<<(DFEAT * DFEAT + 255) / 256, 256, 0, stream>>>(W1, Whi1, Wlo1, DFEAT * DFEAT);
    k_wsplit<<<(DFEAT * DFEAT + 255) / 256, 256, 0, stream>>>(W2, Whi2, Wlo2, DFEAT * DFEAT);

    // degrees -> dinv
    hipMemsetAsync(dinv, 0, (size_t)N * 4, stream);
    k_deg<<<(E + 255) / 256, 256, 0, stream>>>(dst, dinv, E);
    k_dinv<<<(N + 255) / 256, 256, 0, stream>>>(dinv, N);

    // ---- layer 1 ----
    k_gemm_bf16x3<<<(N + 15) / 16, 256, 0, stream>>>(x, Whi1, Wlo1, bufH, N);
    hipMemsetAsync(bufA, 0, nd_bytes, stream);
    k_scatter<<<(E + 7) / 8, 256, 0, stream>>>(bufH, src, dst, dinv, bufA, E);
    k_fuse<<<(N * 32 + 255) / 256, 256, 0, stream>>>(bufA, bufH, dinv, b1, N);

    // ---- layer 2 ----
    k_gemm_bf16x3<<<(N + 15) / 16, 256, 0, stream>>>(bufA, Whi2, Wlo2, bufH, N);
    hipMemsetAsync(bufA, 0, nd_bytes, stream);
    k_scatter<<<(E + 7) / 8, 256, 0, stream>>>(bufH, src, dst, dinv, bufA, E);
    k_fuse<<<(N * 32 + 255) / 256, 256, 0, stream>>>(bufA, bufH, dinv, b2, N);

    // ---- mean pool ----
    hipMemsetAsync(out, 0, (size_t)out_size * 4, stream);
    hipMemsetAsync(cnt, 0, (size_t)G * 4, stream);
    k_pool<<<(N * 32 + 255) / 256, 256, 0, stream>>>(bufA, batch, out, cnt, N);
    k_final<<<(G * DFEAT + 255) / 256, 256, 0, stream>>>(out, cnt, G * DFEAT);
}
